// DoubleImplicitGenerator3d_45947560133147
// MI455X (gfx1250) — matrix operand order repaired and checked by reading, not compiled, not run on hardware
//
#include <hip/hip_runtime.h>
#include <math.h>

#define HID  256
#define NSEG 19
#define CCH  23            // 19 seg + 3 rgb + 1 sigma
#define SS   12
#define IMG  64
#define RAYS (IMG*IMG)     // 4096
#define BB   2
#define NPB  (RAYS*SS)     // 49152 points per batch
#define MT   64            // rows (points) per workgroup
#define STR  264           // LDS row stride in bf16 elems (16B-aligned rows)
#define PI_F 3.14159265358979323846f

typedef __attribute__((ext_vector_type(16))) __bf16         v16bf;
typedef __attribute__((ext_vector_type(8)))  float          v8f;
typedef __attribute__((ext_vector_type(8)))  unsigned short u16x8;

union Frag { v16bf v; u16x8 h[2]; };

__device__ __forceinline__ unsigned short f2bf(float f) {
  unsigned u = __float_as_uint(f);
  u += 0x7FFFu + ((u >> 16) & 1u);
  return (unsigned short)(u >> 16);
}
__device__ __forceinline__ float bf2f(unsigned short s) {
  return __uint_as_float(((unsigned)s) << 16);
}

// ---------------------------------------------------------------------------
// Mapping networks: z -> (freq*15+30, phase). blockIdx: 0,1 geo b0/b1; 2,3 app.
// ---------------------------------------------------------------------------
__global__ void map_kernel(const float* zg, const float* za,
                           const float* gW, const float* gb, const float* gWo, const float* gbo,
                           const float* aW, const float* ab, const float* aWo, const float* abo,
                           float* fg, float* pg, float* fa, float* pa) {
  int blk = blockIdx.x;
  int geo = blk < 2;
  int b = blk & 1;
  const float* z  = geo ? zg  : za;
  const float* W  = geo ? gW  : aW;
  const float* bs = geo ? gb  : ab;
  const float* Wo = geo ? gWo : aWo;
  const float* bo = geo ? gbo : abo;
  int OUTD = geo ? 4096 : 512;
  float* fdst = geo ? fg : fa;
  float* pdst = geo ? pg : pa;

  __shared__ float h[HID], hn[HID];
  int t = threadIdx.x;
  h[t] = z[b * HID + t];
  __syncthreads();
  for (int i = 0; i < 3; ++i) {
    float v = bs[i * HID + t];
    const float* Wi = W + i * HID * HID + t;
    for (int k = 0; k < HID; ++k) v += h[k] * Wi[k * HID];
    hn[t] = v > 0.f ? v : 0.2f * v;     // leaky_relu 0.2
    __syncthreads();
    h[t] = hn[t];
    __syncthreads();
  }
  int half = OUTD >> 1;
  for (int j = t; j < OUTD; j += HID) {
    float o = bo[j];
    const float* Wj = Wo + j;
    for (int k = 0; k < HID; ++k) o += h[k] * Wj[k * OUTD];
    if (j < half) fdst[b * half + j]        = o * 15.f + 30.f;
    else          pdst[b * half + j - half] = o;
  }
}

// ---------------------------------------------------------------------------
// Weight convert+transpose to bf16: Wt[n][k] = W[k][n]. 2048 blocks x 256 thr.
// blocks 0..1791: Wh layer w=blk>>8 ; blocks 1792..2047: colorT (rows 3..258)
// ---------------------------------------------------------------------------
__global__ void convw_kernel(const float* Wh, const float* colorW,
                             unsigned short* wt, unsigned short* colorT) {
  int blk = blockIdx.x;
  int w = blk >> 8;
  int n = blk & 255;
  int k = threadIdx.x;
  if (w < 7)
    wt[w * 65536 + n * 256 + k] = f2bf(Wh[w * 65536 + k * 256 + n]);
  else
    colorT[n * 256 + k] = f2bf(colorW[(3 + k) * 256 + n]);
}

// ---------------------------------------------------------------------------
// Ray/point generation. Rm == identity and origin == (0,0,1) for theta=phi=pi/2
// ---------------------------------------------------------------------------
__global__ void genrays_kernel(const float* noise, float* dirs, float* zv, float* pts) {
  int id = blockIdx.x * blockDim.x + threadIdx.x;
  if (id >= BB * RAYS) return;
  int b = id / RAYS, r = id % RAYS;
  float lx = -1.f + 2.f * (float)(r % IMG) / (float)(IMG - 1);
  float ly = -1.f + 2.f * (float)(r / IMG) / (float)(IMG - 1);
  float zc = -1.f / tanf(PI_F * 12.f / 360.f);
  float inv = rsqrtf(lx * lx + ly * ly + zc * zc);
  float dx = lx * inv, dy = ly * inv, dz = zc * inv;
  if (b == 0) { dirs[r * 3 + 0] = dx; dirs[r * 3 + 1] = dy; dirs[r * 3 + 2] = dz; }
  float stepz = (1.12f - 0.88f) / (float)(SS - 1);
  const float* np_ = noise + (size_t)id * SS;
  for (int s = 0; s < SS; ++s) {
    float z = 0.88f + stepz * s + (np_[s] - 0.5f) * stepz;
    zv[(size_t)id * SS + s] = z;
    float* p = pts + ((size_t)id * SS + s) * 3;
    p[0] = dx * z; p[1] = dy * z; p[2] = dz * z + 1.f;
  }
}

// ---------------------------------------------------------------------------
// Fused SIREN MLP: 8 sin-layers (WMMA bf16) + sigma/seg heads + color (WMMA)
// + rgb head. 256 threads = 8 waves; 64 points per block; acts in LDS (bf16).
// ---------------------------------------------------------------------------
__global__ void __launch_bounds__(256)
siren_kernel(const float* pts, const float* dirs,
             const float* fg, const float* pg, const float* fa, const float* pa,
             const float* W0, const float* b0,
             const unsigned short* wt, const float* bh,
             const unsigned short* colorT, const float* colorW, const float* colorb,
             const float* sigW, const float* sigb, const float* segW, const float* segb,
             const float* rgbW, const float* rgbb,
             float* out) {
  __shared__ __align__(16) unsigned short act[MT * STR];
  __shared__ float ptsh[MT * 3];
  __shared__ float dirsh[MT * 3];

  int t = threadIdx.x;
  int b = blockIdx.x / (NPB / MT);
  int tile = blockIdx.x % (NPB / MT);
  int row0 = tile * MT;
  const float* fgp = fg + b * 2048;
  const float* pgp = pg + b * 2048;

  for (int i = t; i < MT * 3; i += 256) {
    ptsh[i] = pts[((size_t)b * NPB + row0) * 3 + i];
    int m = i / 3, c = i % 3;
    dirsh[i] = dirs[((row0 + m) / SS) * 3 + c];
  }
  __syncthreads();

  // first layer: sin(f0 * (p @ W0 + b0) + p0), K=3 via VALU
  for (int i = t; i < MT * HID; i += 256) {
    int m = i >> 8, j = i & 255;
    float v = b0[j] + ptsh[m * 3] * W0[j] + ptsh[m * 3 + 1] * W0[256 + j]
                    + ptsh[m * 3 + 2] * W0[512 + j];
    act[m * STR + j] = f2bf(__sinf(fgp[j] * v + pgp[j]));
  }
  __syncthreads();

  int lane = t & 31, wid = t >> 5;
  int mblk = wid & 3, nhalf = wid >> 2;
  int mrow = mblk * 16 + (lane & 15);
  int koffA = (lane & 16) ? 8 : 0;    // A: lanes 16-31 hold K+8 / K+24
  int koffB = (lane & 16) ? 8 : 0;    // B: lanes 16-31 hold K+8 / K+24
  int rbase = mblk * 16 + ((lane & 16) ? 8 : 0);

  // 7 hidden layers: Y = sin(f * (X @ W + b) + p), 256x256 via WMMA, in-place
  for (int l = 0; l < 7; ++l) {
    Frag afr[8];
    const unsigned short* arow = act + mrow * STR + koffA;
#pragma unroll
    for (int kc = 0; kc < 8; ++kc) {
      afr[kc].h[0] = *(const u16x8*)(arow + kc * 32);
      afr[kc].h[1] = *(const u16x8*)(arow + kc * 32 + 16);
    }
    __syncthreads();   // all A-frags register-resident -> safe in-place writes
    if (l < 6) __builtin_prefetch(wt + (l + 1) * 65536 + t * 128, 0, 1);
    const unsigned short* wl = wt + l * 65536 + koffB;
#pragma unroll
    for (int nn = 0; nn < 8; ++nn) {
      int ncol = (nhalf * 8 + nn) * 16 + (lane & 15);
      const unsigned short* wr = wl + ncol * 256;
      float bias = bh[l * 256 + ncol];
      v8f acc = {bias, bias, bias, bias, bias, bias, bias, bias};
#pragma unroll
      for (int kc = 0; kc < 8; ++kc) {
        Frag bf_;
        bf_.h[0] = *(const u16x8*)(wr + kc * 32);
        bf_.h[1] = *(const u16x8*)(wr + kc * 32 + 16);
        acc = __builtin_amdgcn_wmma_f32_16x16x32_bf16(
            false, afr[kc].v, false, bf_.v, (short)0, acc, false, false);
      }
      float f  = fgp[(l + 1) * 256 + ncol];
      float ph = pgp[(l + 1) * 256 + ncol];
#pragma unroll
      for (int r = 0; r < 8; ++r) {
        act[(rbase + r) * STR + ncol] = f2bf(__sinf(f * acc[r] + ph));
      }
    }
    __syncthreads();
  }

  // Pre-load color-layer A frags (x), then sigma/seg heads read x from LDS
  Frag cafr[8];
  {
    const unsigned short* arow = act + mrow * STR + koffA;
#pragma unroll
    for (int kc = 0; kc < 8; ++kc) {
      cafr[kc].h[0] = *(const u16x8*)(arow + kc * 32);
      cafr[kc].h[1] = *(const u16x8*)(arow + kc * 32 + 16);
    }
  }
  for (int idx = t; idx < MT * 20; idx += 256) {
    int m = idx / 20, c = idx % 20;
    const unsigned short* xr = act + m * STR;
    const float* wp = (c < 19) ? (segW + c) : sigW;
    int wstride = (c < 19) ? 19 : 1;
    float s = (c < 19) ? segb[c] : sigb[0];
    for (int k = 0; k < HID; ++k) s += bf2f(xr[k]) * wp[k * wstride];
    size_t ptI = (size_t)b * NPB + row0 + m;
    out[ptI * CCH + ((c < 19) ? c : 22)] = s;
  }
  __syncthreads();

  // color layer: h = sin(fa * ([dirs,x] @ colorW + cb) + pa), WMMA + dir epilogue
  {
    const float* fap = fa + b * 256;
    const float* pap = pa + b * 256;
    const unsigned short* wl = colorT + koffB;
#pragma unroll
    for (int nn = 0; nn < 8; ++nn) {
      int ncol = (nhalf * 8 + nn) * 16 + (lane & 15);
      const unsigned short* wr = wl + ncol * 256;
      float bias = colorb[ncol];
      v8f acc = {bias, bias, bias, bias, bias, bias, bias, bias};
#pragma unroll
      for (int kc = 0; kc < 8; ++kc) {
        Frag bf_;
        bf_.h[0] = *(const u16x8*)(wr + kc * 32);
        bf_.h[1] = *(const u16x8*)(wr + kc * 32 + 16);
        acc = __builtin_amdgcn_wmma_f32_16x16x32_bf16(
            false, cafr[kc].v, false, bf_.v, (short)0, acc, false, false);
      }
      float f  = fap[ncol];
      float ph = pap[ncol];
      float w0 = colorW[ncol], w1 = colorW[256 + ncol], w2 = colorW[512 + ncol];
#pragma unroll
      for (int r = 0; r < 8; ++r) {
        int row = rbase + r;
        float dc = dirsh[row * 3] * w0 + dirsh[row * 3 + 1] * w1 + dirsh[row * 3 + 2] * w2;
        act[row * STR + ncol] = f2bf(__sinf(f * (acc[r] + dc) + ph));
      }
    }
  }
  __syncthreads();

  // rgb head: sigmoid(h @ rgbW + rgbb)
  for (int idx = t; idx < MT * 3; idx += 256) {
    int m = idx / 3, c = idx % 3;
    const unsigned short* hr = act + m * STR;
    float s = rgbb[c];
    for (int k = 0; k < HID; ++k) s += bf2f(hr[k]) * rgbW[k * 3 + c];
    size_t ptI = (size_t)b * NPB + row0 + m;
    out[ptI * CCH + 19 + c] = 1.f / (1.f + __expf(-s));
  }
}

// ---------------------------------------------------------------------------
// Coarse weights -> inverse-CDF sampling -> fine z + fine points. 1 thr / ray.
// ---------------------------------------------------------------------------
__global__ void finez_kernel(const float* zv, const float* outc, const float* u_fine,
                             const float* dirs, float* fz, float* ptsf) {
  int id = blockIdx.x * blockDim.x + threadIdx.x;
  if (id >= BB * RAYS) return;
  int r = id % RAYS;
  float zl[SS], w[SS];
  for (int s = 0; s < SS; ++s) zl[s] = zv[(size_t)id * SS + s];
  float T = 1.f;
  for (int s = 0; s < SS; ++s) {
    float d  = (s < SS - 1) ? (zl[s + 1] - zl[s]) : 1e10f;
    float sg = fmaxf(outc[((size_t)id * SS + s) * CCH + 22], 0.f);
    float a  = 1.f - __expf(-d * sg);
    w[s] = a * T + 1e-5f;
    T *= (1.f - a + 1e-10f);
  }
  float cdf[11]; cdf[0] = 0.f;
  float sum = 0.f;
  for (int i = 0; i < 10; ++i) sum += w[i + 1] + 1e-5f;
  float run = 0.f;
  for (int i = 0; i < 10; ++i) { run += (w[i + 1] + 1e-5f) / sum; cdf[i + 1] = run; }
  float zm[11];
  for (int i = 0; i < 11; ++i) zm[i] = 0.5f * (zl[i] + zl[i + 1 < SS ? i + 1 : i]);
  for (int i = 0; i < 11; ++i) zm[i] = 0.5f * (zl[i] + zl[i + 1]);
  float dx = dirs[r * 3], dy = dirs[r * 3 + 1], dz = dirs[r * 3 + 2];
  for (int s = 0; s < SS; ++s) {
    float u = u_fine[(size_t)id * SS + s];
    int ind = 0;
    for (int j = 0; j < 11; ++j) ind += (cdf[j] <= u) ? 1 : 0;   // searchsorted right
    int below = ind - 1 > 0 ? ind - 1 : 0;
    int above = ind < 10 ? ind : 10;
    float c0 = cdf[below], c1 = cdf[above];
    float den = (c1 - c0 < 1e-5f) ? 1.f : (c1 - c0);
    float f = zm[below] + (u - c0) / den * (zm[above] - zm[below]);
    fz[(size_t)id * SS + s] = f;
    float* p = ptsf + ((size_t)id * SS + s) * 3;
    p[0] = dx * f; p[1] = dy * f; p[2] = dz * f + 1.f;
  }
}

// ---------------------------------------------------------------------------
// Merge fine+coarse, sort 24 samples by z, integrate, softmax(seg), write out.
// ---------------------------------------------------------------------------
__global__ void composite_kernel(const float* outf, const float* outc,
                                 const float* fz, const float* zv, float* outp) {
  int id = blockIdx.x * blockDim.x + threadIdx.x;
  if (id < 4) outp[(size_t)BB * 22 * RAYS + id] = PI_F * 0.5f;   // [phi, theta] per b
  if (id >= BB * RAYS) return;
  int b = id / RAYS, r = id % RAYS;
  float z[24]; int srt[24];
  for (int s = 0; s < SS; ++s) {
    z[s]      = fz[(size_t)id * SS + s];
    z[SS + s] = zv[(size_t)id * SS + s];
  }
  for (int i = 0; i < 24; ++i) srt[i] = i;
  for (int i = 1; i < 24; ++i) {
    int key = srt[i]; float kz = z[key]; int j = i - 1;
    while (j >= 0 && z[srt[j]] > kz) { srt[j + 1] = srt[j]; --j; }
    srt[j + 1] = key;
  }
  float acc[22];
  for (int c = 0; c < 22; ++c) acc[c] = 0.f;
  float T = 1.f;
  for (int i = 0; i < 24; ++i) {
    int s = srt[i];
    float d = (i < 23) ? (z[srt[i + 1]] - z[s]) : 1e10f;
    const float* samp = (s < SS) ? (outf + ((size_t)id * SS + s) * CCH)
                                 : (outc + ((size_t)id * SS + (s - SS)) * CCH);
    float sg = fmaxf(samp[22], 0.f);
    float a = 1.f - __expf(-d * sg);
    float wgt = a * T;
    T *= (1.f - a + 1e-10f);
    for (int c = 0; c < 22; ++c) acc[c] += wgt * samp[c];
  }
  float mx = acc[0];
  for (int c = 1; c < 19; ++c) mx = fmaxf(mx, acc[c]);
  float ex[19]; float se = 0.f;
  for (int c = 0; c < 19; ++c) { ex[c] = __expf(acc[c] - mx); se += ex[c]; }
  for (int c = 0; c < 19; ++c)
    outp[((size_t)(b * 22 + c)) * RAYS + r] = (ex[c] / se) * 2.f - 1.f;
  for (int c = 19; c < 22; ++c)
    outp[((size_t)(b * 22 + c)) * RAYS + r] = acc[c] * 2.f - 1.f;
}

// ---------------------------------------------------------------------------
// Workspace layout (float offsets)
// ---------------------------------------------------------------------------
#define OFS_FG   0
#define OFS_PG   4096
#define OFS_FA   8192
#define OFS_PA   8704
#define OFS_DIRS 9216
#define OFS_ZC   21504
#define OFS_PTSC 119808
#define OFS_OUTC 414720
#define OFS_FZ   2675712
#define OFS_PTSF 2774016
#define OFS_OUTF 3068928
#define OFS_WT   5329920   // bf16 region: 7*65536 + 65536 ushorts

extern "C" void kernel_launch(void* const* d_in, const int* in_sizes, int n_in,
                              void* d_out, int out_size, void* d_ws, size_t ws_size,
                              hipStream_t stream) {
  const float* z_geo   = (const float*)d_in[0];
  const float* z_app   = (const float*)d_in[1];
  const float* gmW     = (const float*)d_in[2];
  const float* gmb     = (const float*)d_in[3];
  const float* gmWo    = (const float*)d_in[4];
  const float* gmbo    = (const float*)d_in[5];
  const float* amW     = (const float*)d_in[6];
  const float* amb     = (const float*)d_in[7];
  const float* amWo    = (const float*)d_in[8];
  const float* ambo    = (const float*)d_in[9];
  const float* W0      = (const float*)d_in[10];
  const float* b0      = (const float*)d_in[11];
  const float* Wh      = (const float*)d_in[12];
  const float* bh      = (const float*)d_in[13];
  const float* sigma_W = (const float*)d_in[14];
  const float* sigma_b = (const float*)d_in[15];
  const float* seg_W   = (const float*)d_in[16];
  const float* seg_b   = (const float*)d_in[17];
  const float* color_W = (const float*)d_in[18];
  const float* color_b = (const float*)d_in[19];
  const float* rgb_W   = (const float*)d_in[20];
  const float* rgb_b   = (const float*)d_in[21];
  const float* noise   = (const float*)d_in[22];
  const float* u_fine  = (const float*)d_in[23];

  float* W = (float*)d_ws;
  float* fg   = W + OFS_FG;
  float* pg   = W + OFS_PG;
  float* fa   = W + OFS_FA;
  float* pa   = W + OFS_PA;
  float* dirs = W + OFS_DIRS;
  float* zvc  = W + OFS_ZC;
  float* ptsc = W + OFS_PTSC;
  float* outc = W + OFS_OUTC;
  float* fzv  = W + OFS_FZ;
  float* ptsf = W + OFS_PTSF;
  float* outf = W + OFS_OUTF;
  unsigned short* wt     = (unsigned short*)(W + OFS_WT);
  unsigned short* colorT = wt + 7 * 65536;

  map_kernel<<<4, 256, 0, stream>>>(z_geo, z_app, gmW, gmb, gmWo, gmbo,
                                    amW, amb, amWo, ambo, fg, pg, fa, pa);
  convw_kernel<<<2048, 256, 0, stream>>>(Wh, color_W, wt, colorT);
  genrays_kernel<<<(BB * RAYS + 255) / 256, 256, 0, stream>>>(noise, dirs, zvc, ptsc);

  siren_kernel<<<BB * (NPB / MT), 256, 0, stream>>>(
      ptsc, dirs, fg, pg, fa, pa, W0, b0, wt, bh, colorT, color_W, color_b,
      sigma_W, sigma_b, seg_W, seg_b, rgb_W, rgb_b, outc);

  finez_kernel<<<(BB * RAYS + 255) / 256, 256, 0, stream>>>(zvc, outc, u_fine, dirs, fzv, ptsf);

  siren_kernel<<<BB * (NPB / MT), 256, 0, stream>>>(
      ptsf, dirs, fg, pg, fa, pa, W0, b0, wt, bh, colorT, color_W, color_b,
      sigma_W, sigma_b, seg_W, seg_b, rgb_W, rgb_b, outf);

  composite_kernel<<<(BB * RAYS + 255) / 256, 256, 0, stream>>>(
      outf, outc, fzv, zvc, (float*)d_out);
}
